// meta_surrogate_q_40020505264693
// MI455X (gfx1250) — matrix operand order repaired and checked by reading, not compiled, not run on hardware
//
#include <hip/hip_runtime.h>
#include <hip/hip_bf16.h>
#include <math.h>
#include <stdint.h>

// ---------------------------------------------------------------------------
// Types for CDNA5 WMMA
// ---------------------------------------------------------------------------
typedef __attribute__((ext_vector_type(16))) _Float16 v16h;
typedef __attribute__((ext_vector_type(8)))  _Float16 v8h;
typedef __attribute__((ext_vector_type(8)))  float    v8f;
typedef __attribute__((ext_vector_type(4)))  float    v4f;

// Global (address_space(1)) pointer views so the backend emits global_load_*
typedef const __attribute__((address_space(1))) _Float16* gh_p;
typedef const __attribute__((address_space(1))) float*    gf_p;
typedef const __attribute__((address_space(1))) v16h*     gv16_p;
typedef const __attribute__((address_space(1))) v4f*      gv4f_p;

#if __has_builtin(__builtin_amdgcn_tanhf)
#define TANHF(v) __builtin_amdgcn_tanhf(v)   // gfx1250 v_tanh_f32 (TRANS op)
#else
#define TANHF(v) tanhf(v)
#endif

// ---------------------------------------------------------------------------
// Parameter table: d_in layout (jax tree-flatten, dict keys sorted):
//   0..10 : h1,z1,h2,z2,sigma,aold,amid,qlast,yFull,aFull,rFull   [N] f32
//   11    : length   12: dt                                        [N] f32
//   13 +  net*14 + leaf : params
//     nets sorted: Wslot(0),a(1),dq1(2),dq2(3),dq3(4),dq4(5),dq5(6),dq6(7),
//                  q(8),r(9),v(10)
//     leaves: K_w,K_b,dec0_w,dec0_b,dec1_w,dec1_b,enc0_w,enc0_b,
//             enc1_w,enc1_b,in_w,in_b,out_w,out_b
// ---------------------------------------------------------------------------
#define NPTR 167
struct PTab { const float* p[NPTR]; };

__device__ __constant__ static const int kNin[11] = {2,4,2,4,4,4,4,2,7,3,2};

#define NET_HALF_STRIDE 22528   // 44 frags * 512 halves
#define FLOAT_SEC_OFF   (11 * NET_HALF_STRIDE * 2)  // bytes: 495,616 (16B aligned)
#define NETF_STRIDE     512     // floats per net in float section

// ---------------------------------------------------------------------------
// Prep kernel: swizzle f32 weights -> f16 WMMA B-fragments in ws, plus an f32
// section with biases / out weights.
//   blocks 0..483   : frag (net = b/44, frag = b%44)
//   blocks 484..494 : per-net bias/out float section
// Fragment layout per net: frag 0..3 = in-layer (K padded to 32), ntile 0..3
//                          frag 4+   = (layer-1)*8 + kstep*4 + ntile
// Within a frag: half index = lane*16 + e,
//   B element: column n = ntile*16 + (lane&15), K = kstep*32 + (e>>3)*16 + (lane>>4)*8 + (e&7)
// ---------------------------------------------------------------------------
__global__ void prep_kernel(PTab pt, _Float16* __restrict__ wsw,
                            float* __restrict__ wsf) {
    const int WLEAF[6] = {10, 6, 8, 0, 2, 4};
    const int BLEAF[6] = {11, 7, 9, 1, 3, 5};

    if (blockIdx.x < 484) {
        const int net = blockIdx.x / 44;
        const int f   = blockIdx.x % 44;
        int layer, kt, nt;
        if (f < 4) { layer = 0; kt = 0; nt = f; }
        else { int g = f - 4; layer = 1 + (g >> 3); kt = (g >> 2) & 1; nt = g & 3; }
        const float* w = pt.p[13 + net * 14 + WLEAF[layer]];
        _Float16* dst  = wsw + net * NET_HALF_STRIDE + f * 512;
        const int Kin  = 2 * kNin[net];
        for (int t = threadIdx.x; t < 512; t += blockDim.x) {
            const int lanev = t >> 4, e = t & 15;
            const int K = kt * 32 + ((e >> 3) << 4) + ((lanev >> 4) << 3) + (e & 7);
            const int n = nt * 16 + (lanev & 15);
            float val;
            if (layer == 0) val = (K < Kin) ? w[K * 64 + n] : 0.0f;
            else            val = w[K * 64 + n];
            dst[t] = (_Float16)val;
        }
    } else {
        const int net = blockIdx.x - 484;
        float* nf = wsf + net * NETF_STRIDE;
        for (int t = threadIdx.x; t < 449; t += blockDim.x) {
            float val;
            if (t < 384) {            // 6 bias vectors of 64
                const int L = t >> 6, c = t & 63;
                val = pt.p[13 + net * 14 + BLEAF[L]][c];
            } else if (t < 448) {     // out_w[64]
                val = pt.p[13 + net * 14 + 12][t - 384];
            } else {                  // out_b
                val = pt.p[13 + net * 14 + 13][0];
            }
            nf[t] = val;
        }
    }
}

// ---------------------------------------------------------------------------
// One full net forward for the 16 samples held by this wave.
//   wnet_g : f16 fragment base for this net (global)
//   nf_g   : f32 section for this net (biases, out_w, out_b) (global)
// LDS staging tile is declared *inside* this function so the address space is
// visible to the backend -> real DS instructions (wave-private region, no
// barriers needed: DS ops are in-order within a wave).
// A-fragment layout (16-bit A 16x32, ISA 7.12.2):
//   lane l holds row m=l&15; element e -> K = (e>=8?16:0) + (l>=16?8:0) + (e&7)
// C/D layout: VGPR r, lane l -> row M = r + 8*(l>>4), col N = (l&15) + 16*nt
// ---------------------------------------------------------------------------
__device__ __attribute__((noinline))
float net_fwd(const _Float16* __restrict__ wnet_g,
              const float* __restrict__ nf_g, int nin, const float* x) {
    __shared__ _Float16 stage[8][16 * 72];

    const int tid  = __builtin_amdgcn_workitem_id_x();
    const int wave = (tid >> 5) & 7;
    const int lane = tid & 31;
    const int m    = lane & 15;
    const int hi   = lane >> 4;
    _Float16* st   = &stage[wave][0];

    gh_p wg = (gh_p)wnet_g;
    gf_p nf = (gf_p)nf_g;

    // ---- input layer: build A fragment from positional encoding ----
    v16h a0;
#pragma unroll
    for (int e = 0; e < 16; ++e) {
        const int K = ((e >> 3) << 4) + (hi << 3) + (e & 7);
        float val = 0.0f;
        if (K < nin)          val = __ocml_cos_f32(x[K]);
        else if (K < 2 * nin) val = __ocml_sin_f32(x[K - nin]);
        a0[e] = (_Float16)val;
    }
    v8f acc[4];
#pragma unroll
    for (int nt = 0; nt < 4; ++nt) {
        v8f c = {};
        v16h b = *(gv16_p)(wg + nt * 512 + lane * 16);
        acc[nt] = __builtin_amdgcn_wmma_f32_16x16x32_f16(
            false, a0, false, b, (short)0, c, false, false);
    }
    // bias (layer 0, linear), stage to LDS as f16
#pragma unroll
    for (int nt = 0; nt < 4; ++nt) {
        const float bv = nf[0 * 64 + nt * 16 + m];
#pragma unroll
        for (int r = 0; r < 8; ++r) {
            const float vv = acc[nt][r] + bv;
            st[(r + hi * 8) * 72 + nt * 16 + m] = (_Float16)vv;
        }
    }

    // ---- five 64x64 layers: enc0,enc1,K,dec0,dec1 ----
    const bool DO_TANH[6] = {false, true, true, false, true, true};
#pragma unroll
    for (int L = 1; L <= 5; ++L) {
        // load both A fragments (kstep 0/1) from staged activations
        v16h A[2];
#pragma unroll
        for (int kt = 0; kt < 2; ++kt) {
            const int cb = kt * 32 + hi * 8;
            v8h lo = *(const v8h*)(st + m * 72 + cb);
            v8h hg = *(const v8h*)(st + m * 72 + cb + 16);
            v16h a;
#pragma unroll
            for (int j = 0; j < 8; ++j) { a[j] = lo[j]; a[8 + j] = hg[j]; }
            A[kt] = a;
        }
        gh_p fb = wg + (4 + (L - 1) * 8) * 512;
#pragma unroll
        for (int nt = 0; nt < 4; ++nt) {
            v8f c = {};
            v16h b0 = *(gv16_p)(fb + nt * 512 + lane * 16);
            c = __builtin_amdgcn_wmma_f32_16x16x32_f16(
                false, A[0], false, b0, (short)0, c, false, false);
            v16h b1 = *(gv16_p)(fb + (4 + nt) * 512 + lane * 16);
            c = __builtin_amdgcn_wmma_f32_16x16x32_f16(
                false, A[1], false, b1, (short)0, c, false, false);
            acc[nt] = c;
        }
        // bias (+tanh via v_tanh_f32), restage
#pragma unroll
        for (int nt = 0; nt < 4; ++nt) {
            const float bv = nf[L * 64 + nt * 16 + m];
#pragma unroll
            for (int r = 0; r < 8; ++r) {
                float vv = acc[nt][r] + bv;
                if (DO_TANH[L]) vv = TANHF(vv);
                st[(r + hi * 8) * 72 + nt * 16 + m] = (_Float16)vv;
            }
        }
    }

    // ---- out layer: per-lane dot over staged row (vectorized weight loads) --
    gf_p ow = nf + 384;
    float s = nf[448];
#pragma unroll
    for (int kc = 0; kc < 8; ++kc) {
        v8h hv = *(const v8h*)(st + m * 72 + kc * 8);
        v4f w0 = *(gv4f_p)(ow + kc * 8);
        v4f w1 = *(gv4f_p)(ow + kc * 8 + 4);
#pragma unroll
        for (int j = 0; j < 4; ++j) s += (float)hv[j] * w0[j];
#pragma unroll
        for (int j = 0; j < 4; ++j) s += (float)hv[4 + j] * w1[j];
    }
    return s;
}

// ---------------------------------------------------------------------------
// Fused main kernel: 8 waves/block, 16 samples/wave, 13 net forwards/sample.
// ---------------------------------------------------------------------------
__global__ __launch_bounds__(256)
void fused_kernel(const float* __restrict__ h1p, const float* __restrict__ z1p,
                  const float* __restrict__ h2p, const float* __restrict__ z2p,
                  const float* __restrict__ sgp, const float* __restrict__ aop,
                  const float* __restrict__ amp, const float* __restrict__ qlp,
                  const float* __restrict__ yFp, const float* __restrict__ aFp,
                  const float* __restrict__ rFp, const float* __restrict__ lnp,
                  const float* __restrict__ dtp,
                  const _Float16* __restrict__ wsw, const float* __restrict__ wsf,
                  float* __restrict__ out, int n) {
    const int tid  = threadIdx.x;
    const int wave = tid >> 5;
    const int lane = tid & 31;
    const int m    = lane & 15;
    const int sbase = (blockIdx.x * 8 + wave) * 16;
    int si = sbase + m;
    if (si >= n) si = n - 1;            // clamp loads; keep EXEC uniform for WMMA

    const float h1 = h1p[si], z1 = z1p[si], h2 = h2p[si], z2 = z2p[si];
    const float sigma = sgp[si], aold = aop[si], amid = amp[si], qlast = qlp[si];
    const float yF = yFp[si], aF = aFp[si], rF = rFp[si];
    const float len = lnp[si], dt = dtp[si];

    const float y1 = h1 - z1, y2 = h2 - z2, ymid = 0.5f * (y1 + y2);

#define WNET(i) (wsw + (i) * NET_HALF_STRIDE)
#define NETF(i) (wsf + (i) * NETF_STRIDE)
    float x[7];

    x[0] = y1; x[1] = yF;
    const float W1 = net_fwd(WNET(0), NETF(0), 2, x);
    x[0] = y2;
    const float W2 = net_fwd(WNET(0), NETF(0), 2, x);
    x[0] = y1; x[1] = W1; x[2] = yF; x[3] = aF;
    const float a1 = net_fwd(WNET(1), NETF(1), 4, x);
    x[0] = y2; x[1] = W2;
    const float a2 = net_fwd(WNET(1), NETF(1), 4, x);
    x[0] = ymid; x[1] = yF; x[2] = rF;              // rwtd == rmid (r1 is dead)
    const float rmid = net_fwd(WNET(9), NETF(9), 3, x);
    x[0] = qlast; x[1] = amid;
    const float vv = net_fwd(WNET(10), NETF(10), 2, x);
    x[0] = vv; x[1] = rmid;
    const float dq1 = dt * net_fwd(WNET(2), NETF(2), 2, x);
    x[0] = h1; x[1] = h2; x[2] = amid; x[3] = len;  // awtd == amid
    const float dq2 = dt * 9.8f * net_fwd(WNET(3), NETF(3), 4, x);
    x[0] = vv; x[1] = sigma; x[2] = amid; x[3] = aold;
    const float dq3 = 2.0f * sigma * net_fwd(WNET(4), NETF(4), 4, x);
    x[0] = vv; x[1] = a1; x[2] = a2; x[3] = len;
    const float dq4 = dt * sigma * net_fwd(WNET(5), NETF(5), 4, x);
    x[0] = a1; x[1] = qlast; x[2] = amid; x[3] = a2;
    const float dq5 = 2.0f * len * dt * net_fwd(WNET(6), NETF(6), 4, x);
    x[0] = qlast; x[1] = dt;
    const float dq6 = dt * 2.5f * vv / len * net_fwd(WNET(7), NETF(7), 2, x);
    x[0] = qlast; x[1] = dq1; x[2] = dq2; x[3] = dq3; x[4] = dq4; x[5] = dq5; x[6] = dq6;
    const float q = net_fwd(WNET(8), NETF(8), 7, x);

    if (lane < 16 && (sbase + m) < n) out[sbase + m] = q;
#undef WNET
#undef NETF
}

// ---------------------------------------------------------------------------
extern "C" void kernel_launch(void* const* d_in, const int* in_sizes, int n_in,
                              void* d_out, int out_size, void* d_ws, size_t ws_size,
                              hipStream_t stream) {
    PTab pt{};
    const int np = (n_in < NPTR) ? n_in : NPTR;
    for (int i = 0; i < np; ++i) pt.p[i] = (const float*)d_in[i];

    _Float16* wsw = (_Float16*)d_ws;
    float*    wsf = (float*)((char*)d_ws + FLOAT_SEC_OFF);
    const int n   = in_sizes[0];

    // 1) swizzle weights into WMMA fragment layout (f16) + bias/out f32 section
    prep_kernel<<<495, 256, 0, stream>>>(pt, wsw, wsf);

    // 2) fused forward: 128 samples per block
    const int blocks = (n + 127) / 128;
    fused_kernel<<<blocks, 256, 0, stream>>>(
        pt.p[0], pt.p[1], pt.p[2], pt.p[3], pt.p[4], pt.p[5], pt.p[6], pt.p[7],
        pt.p[8], pt.p[9], pt.p[10], pt.p[11], pt.p[12],
        wsw, wsf, (float*)d_out, n);
}
